// ReadoutResNet_46608985096478
// MI455X (gfx1250) — matrix operand order repaired and checked by reading, not compiled, not run on hardware
//
#include <hip/hip_runtime.h>

typedef __attribute__((ext_vector_type(16))) __bf16 v16bf;
typedef __attribute__((ext_vector_type(8)))  __bf16 v8bf;
typedef __attribute__((ext_vector_type(8)))  float  v8f;

#define D_MODEL 512
#define RD_     48
#define B_      2048
#define NPOS    36                 // 6*6 conv output positions
#define M_TOTAL (B_ * NPOS)        // 73728
#define KTOT    2048               // 4 * 512
#define KSTEPS  (KTOT / 32)        // 64
#define NT_ALL  (D_MODEL / 16)     // 32 N-tiles
#define BM      64                 // rows per block (4 M-tiles)
#define GS_STRIDE (D_MODEL + 4)    // padded to kill LDS bank conflicts

__device__ __forceinline__ float gelu_exact(float x) {
    return 0.5f * x * (1.0f + erff(x * 0.7071067811865476f));
}

// ---------------------------------------------------------------------------
// Kernel 1: repack conv_w (O=512, I=512, 2, 2) f32 -> bf16 WMMA B-fragments.
// Layout: Bpack[ks][nt][lane][i] ; 16-bit B (32x16): N = nt*16 + lane%16,
// K = ks*32 + (lane/16)*16 + i   (i = 0..15)                   (contiguous k; A uses same)
// GEMM K ordering: k = seg*512 + c, seg = dy*2+dx (matches A token order).
// conv_w index [o][c][dy][dx] = o*2048 + c*4 + seg.
// ---------------------------------------------------------------------------
__global__ __launch_bounds__(256) void pack_w_kernel(const float* __restrict__ conv_w,
                                                     __bf16* __restrict__ Bpack) {
    int idx = blockIdx.x * 256 + threadIdx.x;          // 0 .. 64*32*32-1
    if (idx >= KSTEPS * NT_ALL * 32) return;
    int lane = idx & 31;
    int nt   = (idx >> 5) & 31;
    int ks   = idx >> 10;
    int n     = nt * 16 + (lane & 15);
    int kbase = ks * 32 + ((lane >> 4) << 4);
    __bf16* dst = Bpack + (size_t)idx * 16;
#pragma unroll
    for (int i = 0; i < 16; ++i) {
        int k = kbase + i;
        int c = k & 511;
        int s = k >> 9;                                 // seg = dy*2+dx
        dst[i] = (__bf16)conv_w[(size_t)n * 2048 + (size_t)c * 4 + s];
    }
}

// ---------------------------------------------------------------------------
// Kernel 2: fused conv(2x2 s2) GEMM [M=73728,N=512,K=2048] via WMMA bf16 +
// bias + GELU + fused 512->48 projection + GELU.
// Block: 64 rows (4 M-tiles), 8 waves; wave = 4 M-tiles x 4 N-tiles (span 64)
// = 16 accumulator tiles, so each B-fragment feeds 4 WMMAs (L2 traffic /4).
// ---------------------------------------------------------------------------
__global__ __launch_bounds__(256) void conv_gemm_kernel(const float*  __restrict__ X,
                                                        const float*  __restrict__ conv_b,
                                                        const float*  __restrict__ dr_w,
                                                        const float*  __restrict__ dr_b,
                                                        const __bf16* __restrict__ Bpack,
                                                        float* __restrict__ x2) {
    __shared__ __attribute__((aligned(16))) __bf16 As[2][BM * 32];       // 2 x 4 KB
    __shared__ __attribute__((aligned(16))) float  Gs[16 * GS_STRIDE];   // 33 KB

    const int tid  = threadIdx.x;
    const int wave = tid >> 5;          // 0..7
    const int lane = tid & 31;
    const int m0   = blockIdx.x * BM;

    // --- A-tile stage role: thread t loads 8 floats of row r = t/4 ---------
    const int r  = tid >> 2;            // 0..63
    const int c8 = (tid & 3) * 8;       // 0,8,16,24
    const int gm = m0 + r;
    const int b  = gm / NPOS;
    const int pq = gm - b * NPOS;
    const int p  = pq / 6;
    const int q  = pq - p * 6;
    // token (2p,2q) base; seg offsets: (dy*12 + dx)*512
    const float* base00 = X + (((size_t)b * 144) + (size_t)(2 * p) * 12 + (size_t)(2 * q)) * D_MODEL + c8;

    v8f acc[4][4];
#pragma unroll
    for (int mt = 0; mt < 4; ++mt)
#pragma unroll
        for (int nt = 0; nt < 4; ++nt) acc[mt][nt] = (v8f)0.0f;

    const int n0 = wave * 64;
    // A fragment addressing (16-bit A): lane<16 -> K {0..15},
    // lane>=16 -> K {16..31}; M = lane%16.
    const int arow = lane & 15;
    const int koff = (lane & 16) ? 16 : 0;

    for (int ks = 0; ks < KSTEPS; ++ks) {
        const int k0  = ks * 32;
        const int seg = k0 >> 9;
        const int c0  = k0 & 511;
        const int segoff = ((seg >> 1) * 12 + (seg & 1)) * D_MODEL;
        const float4 f0 = *(const float4*)(base00 + segoff + c0);
        const float4 f1 = *(const float4*)(base00 + segoff + c0 + 4);

        const int buf = ks & 1;
        union { v8bf v; } pk;
        pk.v[0] = (__bf16)f0.x; pk.v[1] = (__bf16)f0.y;
        pk.v[2] = (__bf16)f0.z; pk.v[3] = (__bf16)f0.w;
        pk.v[4] = (__bf16)f1.x; pk.v[5] = (__bf16)f1.y;
        pk.v[6] = (__bf16)f1.z; pk.v[7] = (__bf16)f1.w;
        *(v8bf*)&As[buf][r * 32 + c8] = pk.v;
        __syncthreads();                 // includes dscnt wait: ping-pong safe

        union { v16bf v; v8bf h[2]; } au[4];
#pragma unroll
        for (int mt = 0; mt < 4; ++mt) {
            const __bf16* ap = &As[buf][(mt * 16 + arow) * 32 + koff];
            au[mt].h[0] = *(const v8bf*)(ap);
            au[mt].h[1] = *(const v8bf*)(ap + 8);
        }

        const __bf16* bp = Bpack + (((size_t)ks * NT_ALL + (n0 >> 4)) * 32 + lane) * 16;
#pragma unroll
        for (int nt = 0; nt < 4; ++nt) {
            union { v16bf v; v8bf h[2]; } bu;
            const __bf16* src = bp + (size_t)nt * (32 * 16);
            bu.h[0] = *(const v8bf*)(src);
            bu.h[1] = *(const v8bf*)(src + 8);
#pragma unroll
            for (int mt = 0; mt < 4; ++mt) {
                acc[mt][nt] = __builtin_amdgcn_wmma_f32_16x16x32_bf16(
                    false, au[mt].v, false, bu.v, (short)0, acc[mt][nt], false, false);
            }
        }
    }

    // --- epilogue: 4 passes (one per M-tile) through 16x512 LDS buffer ----
    // Fully unrolled so acc[][] indices stay compile-time constant (no
    // dynamic register-array indexing -> no scratch spills).
    // C/D layout: VGPR v -> M = v + (lane<16 ? 0 : 8); N = tile_n0 + lane%16.
    const int mbase = (lane < 16) ? 0 : 8;
#pragma unroll
    for (int mt = 0; mt < 4; ++mt) {
        __syncthreads();                 // previous pass readers done
#pragma unroll
        for (int nt = 0; nt < 4; ++nt) {
            const int n  = n0 + nt * 16 + (lane & 15);
            const float cb = conv_b[n];
#pragma unroll
            for (int v = 0; v < 8; ++v) {
                Gs[(mbase + v) * GS_STRIDE + n] = gelu_exact(acc[mt][nt][v] + cb);
            }
        }
        __syncthreads();

        // fused 512 -> 48 projection + GELU; thread: m = t%16, rd = (t/16)*3+{0..2}
        const int m   = tid & 15;
        const int rd0 = (tid >> 4) * 3;
        const float* g  = Gs + m * GS_STRIDE;
        const float* w0 = dr_w + (size_t)(rd0 + 0) * D_MODEL;
        const float* w1 = dr_w + (size_t)(rd0 + 1) * D_MODEL;
        const float* w2 = dr_w + (size_t)(rd0 + 2) * D_MODEL;
        float a0 = dr_b[rd0 + 0], a1 = dr_b[rd0 + 1], a2 = dr_b[rd0 + 2];
#pragma unroll 4
        for (int c = 0; c < D_MODEL; ++c) {
            const float gv = g[c];
            a0 = fmaf(gv, w0[c], a0);
            a1 = fmaf(gv, w1[c], a1);
            a2 = fmaf(gv, w2[c], a2);
        }
        float* dst = x2 + (size_t)(m0 + mt * 16 + m) * RD_ + rd0;
        dst[0] = gelu_exact(a0);
        dst[1] = gelu_exact(a1);
        dst[2] = gelu_exact(a2);
    }
}

// ---------------------------------------------------------------------------
// Kernel 3: pooling + cycle emb + 16 residual 48x48 blocks, one row/thread.
// ---------------------------------------------------------------------------
__global__ __launch_bounds__(256) void resnet_kernel(const float* __restrict__ x2,
                                                     const float* __restrict__ cycle_emb,
                                                     const int*   __restrict__ basis_idx,
                                                     const int*   __restrict__ cycle_n_p,
                                                     const float* __restrict__ fc1_w,
                                                     const float* __restrict__ fc1_b,
                                                     const float* __restrict__ fc2_w,
                                                     const float* __restrict__ fc2_b,
                                                     const float* __restrict__ out_w,
                                                     const float* __restrict__ out_b,
                                                     float* __restrict__ srow) {
    const int row = blockIdx.x * 256 + threadIdx.x;
    if (row >= B_ * 6) return;
    const int b = row / 6;
    const int j = row - b * 6;
    const int cyc = cycle_n_p[0];
    const float* ce = cycle_emb + (size_t)cyc * RD_;
    const float* xb = x2 + (size_t)b * NPOS * RD_;

    float h[RD_];
    if (basis_idx[b] == 0) {                 // pool over W (q): rows (j*6+q)
#pragma unroll
        for (int rd = 0; rd < RD_; ++rd) {
            float s = 0.0f;
#pragma unroll
            for (int t = 0; t < 6; ++t) s += xb[(j * 6 + t) * RD_ + rd];
            h[rd] = s * (1.0f / 6.0f) + ce[rd];
        }
    } else {                                 // pool over H (p): rows (p*6+j)
#pragma unroll
        for (int rd = 0; rd < RD_; ++rd) {
            float s = 0.0f;
#pragma unroll
            for (int t = 0; t < 6; ++t) s += xb[(t * 6 + j) * RD_ + rd];
            h[rd] = s * (1.0f / 6.0f) + ce[rd];
        }
    }

    for (int l = 0; l < 16; ++l) {
        const float* w1 = fc1_w + (size_t)l * RD_ * RD_;
        const float* b1 = fc1_b + (size_t)l * RD_;
        const float* w2 = fc2_w + (size_t)l * RD_ * RD_;
        const float* b2 = fc2_b + (size_t)l * RD_;
        float t[RD_];
#pragma unroll
        for (int i = 0; i < RD_; ++i) {
            float s = b1[i];
            const float* wr = w1 + (size_t)i * RD_;
#pragma unroll
            for (int k = 0; k < RD_; ++k) s = fmaf(h[k], wr[k], s);
            t[i] = gelu_exact(s);
        }
#pragma unroll
        for (int i = 0; i < RD_; ++i) {
            float s = b2[i];
            const float* wr = w2 + (size_t)i * RD_;
#pragma unroll
            for (int k = 0; k < RD_; ++k) s = fmaf(t[k], wr[k], s);
            h[i] += s;
        }
    }

    float s = out_b[0];
#pragma unroll
    for (int i = 0; i < RD_; ++i) s = fmaf(h[i], out_w[i], s);
    srow[row] = s;
}

// ---------------------------------------------------------------------------
// Kernel 4: mean over K=6 positions -> logits (B,)
// ---------------------------------------------------------------------------
__global__ __launch_bounds__(256) void reduce_kernel(const float* __restrict__ srow,
                                                     float* __restrict__ out) {
    const int b = blockIdx.x * 256 + threadIdx.x;
    if (b >= B_) return;
    float s = 0.0f;
#pragma unroll
    for (int j = 0; j < 6; ++j) s += srow[b * 6 + j];
    out[b] = s * (1.0f / 6.0f);
}

// ---------------------------------------------------------------------------
extern "C" void kernel_launch(void* const* d_in, const int* in_sizes, int n_in,
                              void* d_out, int out_size, void* d_ws, size_t ws_size,
                              hipStream_t stream) {
    const float* X         = (const float*)d_in[0];
    // d_in[1] = P (dead: scatter covers full grid), d_in[14] = stab_ids (identity)
    const float* conv_w    = (const float*)d_in[2];
    const float* conv_b    = (const float*)d_in[3];
    const float* dr_w      = (const float*)d_in[4];
    const float* dr_b      = (const float*)d_in[5];
    const float* cycle_emb = (const float*)d_in[6];
    const float* fc1_w     = (const float*)d_in[7];
    const float* fc1_b     = (const float*)d_in[8];
    const float* fc2_w     = (const float*)d_in[9];
    const float* fc2_b     = (const float*)d_in[10];
    const float* out_w     = (const float*)d_in[11];
    const float* out_b     = (const float*)d_in[12];
    const int*   basis_idx = (const int*)d_in[13];
    const int*   cycle_n   = (const int*)d_in[15];

    char* ws = (char*)d_ws;
    __bf16* Bpack = (__bf16*)ws;                                       // 2 MB
    float*  x2    = (float*)(ws + (2u << 20));                         // 14.2 MB
    float*  srow  = (float*)(ws + (2u << 20) + (size_t)M_TOTAL * RD_ * sizeof(float));

    pack_w_kernel<<<(KSTEPS * NT_ALL * 32 + 255) / 256, 256, 0, stream>>>(conv_w, Bpack);
    conv_gemm_kernel<<<M_TOTAL / BM, 256, 0, stream>>>(X, conv_b, dr_w, dr_b, Bpack, x2);
    resnet_kernel<<<(B_ * 6 + 255) / 256, 256, 0, stream>>>(
        x2, cycle_emb, basis_idx, cycle_n,
        fc1_w, fc1_b, fc2_w, fc2_b, out_w, out_b, srow);
    reduce_kernel<<<(B_ + 255) / 256, 256, 0, stream>>>(srow, (float*)d_out);
}
